// RnnDecoder_85212151153155
// MI455X (gfx1250) — compile-verified, matrix operand order checked by reading, not run on hardware
//
#include <hip/hip_runtime.h>
#include <hip/hip_bf16.h>
#include <cstddef>

// ---- problem constants ----
#define Bz 64
#define Sz 256
#define Hz 1024
#define Ez 256
#define Vz 32000
#define H3 (3 * Hz)
#define HE (Hz + Ez)

typedef __attribute__((ext_vector_type(16))) __bf16 v16bf;
typedef __attribute__((ext_vector_type(8)))  float  v8f;

__device__ __forceinline__ unsigned short bf16_bits(float f) {
  union { __bf16 b; unsigned short u; } c;
  c.b = (__bf16)f;   // RNE hardware convert (v_cvt_pk_bf16_f32 class)
  return c.u;
}

// K-offset mapping inside a 16x32 bf16 fragment:
// lane = lo + 16*hh holds row lo; elems 0..7 <- kb+8*hh+(0..7), elems 8..15 <- kb+16+8*hh+(0..7)
// inverse: ko (0..31) -> hh = (ko>>3)&1 ; ii = (ko&7) | ((ko>>1)&8)
__device__ __forceinline__ void komap(int ko, int& hh, int& ii) {
  hh = (ko >> 3) & 1;
  ii = (ko & 7) | ((ko >> 1) & 8);
}

// On-the-fly convert fragment load from row-major f32 matrix (rows contiguous in K).
__device__ __forceinline__ v16bf load_frag_cvt(const float* __restrict__ base,
                                               int stride, int row0, int kb, int lane) {
  const int lo = lane & 15, hh = lane >> 4;
  const float* p = base + (size_t)(row0 + lo) * (size_t)stride + kb + 8 * hh;
  const float4* q = reinterpret_cast<const float4*>(p);
  float4 x0 = q[0], x1 = q[1], x2 = q[4], x3 = q[5];
  union { v16bf v; __bf16 b[16]; } f;
  f.b[0]  = (__bf16)x0.x; f.b[1]  = (__bf16)x0.y; f.b[2]  = (__bf16)x0.z; f.b[3]  = (__bf16)x0.w;
  f.b[4]  = (__bf16)x1.x; f.b[5]  = (__bf16)x1.y; f.b[6]  = (__bf16)x1.z; f.b[7]  = (__bf16)x1.w;
  f.b[8]  = (__bf16)x2.x; f.b[9]  = (__bf16)x2.y; f.b[10] = (__bf16)x2.z; f.b[11] = (__bf16)x2.w;
  f.b[12] = (__bf16)x3.x; f.b[13] = (__bf16)x3.y; f.b[14] = (__bf16)x3.z; f.b[15] = (__bf16)x3.w;
  return f.v;
}

// Pre-swizzled fragment chunk offset: layout [tile][kk][lane][16]
__device__ __forceinline__ size_t frag_off(int tiles_k, int tile, int kk, int lane) {
  return ((size_t)(tile * tiles_k + kk) * 32 + lane) * 16;
}

__device__ __forceinline__ v16bf load_frag_swz(const unsigned short* __restrict__ base,
                                               int tiles_k, int tile, int kk, int lane) {
  return *reinterpret_cast<const v16bf*>(base + frag_off(tiles_k, tile, kk, lane));
}

__device__ __forceinline__ v8f wmma_bf16(v16bf a, v16bf b, v8f c) {
  return __builtin_amdgcn_wmma_f32_16x16x32_bf16(false, a, false, b, (short)0, c,
                                                 false, false);
}

// ============================================================================
// K0a: convert + swizzle w2_W (1024x1024 f32) into bf16 fragment layout.
// One thread per 16-element chunk: 64 nt * 32 kk * 32 lane = 65536 chunks.
// ============================================================================
__global__ void __launch_bounds__(256)
k0_w2swz(const float* __restrict__ w2W, unsigned short* __restrict__ w2s) {
  const int idx = blockIdx.x * 256 + threadIdx.x;
  const int nt = idx >> 10, kk = (idx >> 5) & 31, lane = idx & 31;
  const int lo = lane & 15, hh = lane >> 4;
  const float* p = w2W + (size_t)(nt * 16 + lo) * Hz + kk * 32 + 8 * hh;
  const float4* q = reinterpret_cast<const float4*>(p);
  float4 x0 = q[0], x1 = q[1], x2 = q[4], x3 = q[5];
  union { uint4 q[2]; __bf16 b[16]; } f;
  f.b[0]  = (__bf16)x0.x; f.b[1]  = (__bf16)x0.y; f.b[2]  = (__bf16)x0.z; f.b[3]  = (__bf16)x0.w;
  f.b[4]  = (__bf16)x1.x; f.b[5]  = (__bf16)x1.y; f.b[6]  = (__bf16)x1.z; f.b[7]  = (__bf16)x1.w;
  f.b[8]  = (__bf16)x2.x; f.b[9]  = (__bf16)x2.y; f.b[10] = (__bf16)x2.z; f.b[11] = (__bf16)x2.w;
  f.b[12] = (__bf16)x3.x; f.b[13] = (__bf16)x3.y; f.b[14] = (__bf16)x3.z; f.b[15] = (__bf16)x3.w;
  uint4* dst = reinterpret_cast<uint4*>(w2s + (size_t)idx * 16);
  dst[0] = f.q[0];
  dst[1] = f.q[1];
}

// ============================================================================
// K0b: convert + swizzle dec_hs (64x1024) into A-fragment layout (4 mt tiles).
// ============================================================================
__global__ void __launch_bounds__(256)
k0_decswz(const float* __restrict__ dec_hs, unsigned short* __restrict__ dswz) {
  const int idx = blockIdx.x * 256 + threadIdx.x;
  const int mt = idx >> 10, kk = (idx >> 5) & 31, lane = idx & 31;
  const int lo = lane & 15, hh = lane >> 4;
  const float* p = dec_hs + (size_t)(mt * 16 + lo) * Hz + kk * 32 + 8 * hh;
  const float4* q = reinterpret_cast<const float4*>(p);
  float4 x0 = q[0], x1 = q[1], x2 = q[4], x3 = q[5];
  union { uint4 q[2]; __bf16 b[16]; } f;
  f.b[0]  = (__bf16)x0.x; f.b[1]  = (__bf16)x0.y; f.b[2]  = (__bf16)x0.z; f.b[3]  = (__bf16)x0.w;
  f.b[4]  = (__bf16)x1.x; f.b[5]  = (__bf16)x1.y; f.b[6]  = (__bf16)x1.z; f.b[7]  = (__bf16)x1.w;
  f.b[8]  = (__bf16)x2.x; f.b[9]  = (__bf16)x2.y; f.b[10] = (__bf16)x2.z; f.b[11] = (__bf16)x2.w;
  f.b[12] = (__bf16)x3.x; f.b[13] = (__bf16)x3.y; f.b[14] = (__bf16)x3.z; f.b[15] = (__bf16)x3.w;
  uint4* dst = reinterpret_cast<uint4*>(dswz + (size_t)idx * 16);
  dst[0] = f.q[0];
  dst[1] = f.q[1];
}

// ============================================================================
// K1: t1[b][h] = dec @ w1^T + w1_b + w2_b.  256 single-wave blocks.
// ============================================================================
__global__ void k1_t1(const unsigned short* __restrict__ dswz, const float* __restrict__ w1W,
                      const float* __restrict__ w1b, const float* __restrict__ w2b,
                      float* __restrict__ t1) {
  const int lane = threadIdx.x;
  const int nt = blockIdx.x % (Hz / 16);
  const int mt = blockIdx.x / (Hz / 16);
  v8f acc = {};
  for (int kk = 0; kk < Hz / 32; ++kk) {
    v16bf a = load_frag_swz(dswz, 32, mt, kk, lane);
    v16bf b = load_frag_cvt(w1W, Hz, nt * 16, kk * 32, lane);
    acc = wmma_bf16(a, b, acc);
  }
  const int lo = lane & 15, hh = lane >> 4;
  const int n = nt * 16 + lo;
  const float bias = w1b[n] + w2b[n];
#pragma unroll
  for (int j = 0; j < 8; ++j)
    t1[(size_t)(mt * 16 + j + 8 * hh) * Hz + n] = acc[j] + bias;
}

// ============================================================================
// K2 (hot): scores[b][s] = v . tanh(t1[b] + enc_row @ w2^T) + v_b
// M=64 rows per block == all 64 batches of one s.  4 A tiles staged in LDS
// (128 KB, pre-swizzled bf16 fragments); 8 waves split the 64 N-tiles;
// each pre-swizzled B fragment feeds 4 WMMAs.  tanh-dot fused per
// accumulator -> feat (64 MB) never materialized.
// ============================================================================
__global__ void __launch_bounds__(256)
k2_scores(const float* __restrict__ enc, const unsigned short* __restrict__ w2s,
          const float* __restrict__ t1, const float* __restrict__ vW,
          const float* __restrict__ vb, float* __restrict__ scores) {
  __shared__ __align__(32) unsigned short aF[Hz / 32][4][32][16];  // 128 KB
  __shared__ float scoreLds[64];

  const int tid = threadIdx.x;
  const int s = blockIdx.x;           // rows r = s*64 + b, b = 0..63
  const int r0 = s * 64;

  if (tid < 64) scoreLds[tid] = 0.0f;

  for (int idx = tid; idx < 64 * Hz; idx += 256) {
    const int m = idx >> 10, k = idx & (Hz - 1);   // m = b
    const float val = enc[(size_t)(r0 + m) * Hz + k];
    const int kk = k >> 5, ko = k & 31;
    int hh, ii;
    komap(ko, hh, ii);
    aF[kk][m >> 4][(m & 15) + 16 * hh][ii] = bf16_bits(val);
  }
  __syncthreads();

  const int wave = tid >> 5, lane = tid & 31;
  const int lo = lane & 15, hh = lane >> 4;

  float sP[4][8];
#pragma unroll
  for (int st = 0; st < 4; ++st)
#pragma unroll
    for (int j = 0; j < 8; ++j) sP[st][j] = 0.0f;

  for (int i = 0; i < 8; ++i) {       // 8 N-tiles per wave (64 total)
    const int nt = wave * 8 + i;
    v8f acc[4] = {{}, {}, {}, {}};
#pragma unroll 4
    for (int kk = 0; kk < Hz / 32; ++kk) {
      v16bf bfr = load_frag_swz(w2s, 32, nt, kk, lane);
#pragma unroll
      for (int st = 0; st < 4; ++st) {
        v16bf a = *reinterpret_cast<const v16bf*>(&aF[kk][st][lane][0]);
        acc[st] = wmma_bf16(a, bfr, acc[st]);
      }
    }
    const int n = nt * 16 + lo;
    const float vn = vW[n];
#pragma unroll
    for (int st = 0; st < 4; ++st)
#pragma unroll
      for (int j = 0; j < 8; ++j) {
        const int b = st * 16 + j + 8 * hh;
        sP[st][j] += tanhf(acc[st][j] + t1[(size_t)b * Hz + n]) * vn;
      }
  }
#pragma unroll
  for (int st = 0; st < 4; ++st)
#pragma unroll
    for (int j = 0; j < 8; ++j)
      atomicAdd(&scoreLds[st * 16 + j + 8 * hh], sP[st][j]);
  __syncthreads();

  if (tid < 64)
    scores[tid * Sz + s] = scoreLds[tid] + vb[0];   // scores[b][s]
}

// ============================================================================
// K3: softmax over S, attn out, context + emb -> GRU input u, written directly
// as swizzled bf16 A-fragments (K=1280 -> 40 kk tiles).
// ============================================================================
__global__ void __launch_bounds__(256)
k3_softmax_ctx(const float* __restrict__ scores, const float* __restrict__ enc,
               const float* __restrict__ emb_table, const int* __restrict__ x,
               unsigned short* __restrict__ uswz, float* __restrict__ out, size_t attn_off) {
  __shared__ float red[Sz];
  __shared__ float attnL[Sz];
  const int b = blockIdx.x, t = threadIdx.x;

  const float sc = scores[b * Sz + t];
  red[t] = sc;
  __syncthreads();
  for (int off = 128; off > 0; off >>= 1) {
    if (t < off) red[t] = fmaxf(red[t], red[t + off]);
    __syncthreads();
  }
  const float mx = red[0];
  __syncthreads();
  const float e = expf(sc - mx);
  red[t] = e;
  __syncthreads();
  for (int off = 128; off > 0; off >>= 1) {
    if (t < off) red[t] += red[t + off];
    __syncthreads();
  }
  const float a = e / red[0];
  attnL[t] = a;
  out[attn_off + (size_t)b * Sz + t] = a;
  __syncthreads();

  float a0 = 0.f, a1 = 0.f, a2 = 0.f, a3 = 0.f;
  for (int s = 0; s < Sz; ++s) {
    const float w = attnL[s];
    const float* er = enc + (size_t)(s * Bz + b) * Hz;
    a0 += w * er[t];
    a1 += w * er[256 + t];
    a2 += w * er[512 + t];
    a3 += w * er[768 + t];
  }
  const int mt = b >> 4, m = b & 15;
  auto put = [&](int kpos, float v) {
    const int kk = kpos >> 5, ko = kpos & 31;
    int hh, ii;
    komap(ko, hh, ii);
    uswz[frag_off(HE / 32, mt, kk, m + 16 * hh) + ii] = bf16_bits(v);
  };
  put(t, a0);
  put(256 + t, a1);
  put(512 + t, a2);
  put(768 + t, a3);
  put(Hz + t, emb_table[(size_t)x[b] * Ez + t]);
}

// ============================================================================
// K4: gi = u @ W_ih^T + b_ih, gh = h @ W_hh^T + b_hh
// One wave per N-tile (192), iterating all 4 M-tiles so each converted
// weight fragment feeds 4 WMMAs. A fragments pre-swizzled bf16.
// ============================================================================
__global__ void k4_gru_gemm(const unsigned short* __restrict__ uswz,
                            const unsigned short* __restrict__ dswz,
                            const float* __restrict__ Wih, const float* __restrict__ Whh,
                            const float* __restrict__ bih, const float* __restrict__ bhh,
                            float* __restrict__ gi, float* __restrict__ gh) {
  const int lane = threadIdx.x;
  const int nt = blockIdx.x;
  const int lo = lane & 15, hh = lane >> 4;
  const int n = nt * 16 + lo;

  v8f ai[4] = {{}, {}, {}, {}};
  for (int kk = 0; kk < HE / 32; ++kk) {
    v16bf bfr = load_frag_cvt(Wih, HE, nt * 16, kk * 32, lane);
#pragma unroll
    for (int mt = 0; mt < 4; ++mt)
      ai[mt] = wmma_bf16(load_frag_swz(uswz, HE / 32, mt, kk, lane), bfr, ai[mt]);
  }
  const float bi = bih[n];
#pragma unroll
  for (int mt = 0; mt < 4; ++mt)
#pragma unroll
    for (int j = 0; j < 8; ++j)
      gi[(size_t)(mt * 16 + j + 8 * hh) * H3 + n] = ai[mt][j] + bi;

  v8f ah[4] = {{}, {}, {}, {}};
  for (int kk = 0; kk < Hz / 32; ++kk) {
    v16bf bfr = load_frag_cvt(Whh, Hz, nt * 16, kk * 32, lane);
#pragma unroll
    for (int mt = 0; mt < 4; ++mt)
      ah[mt] = wmma_bf16(load_frag_swz(dswz, Hz / 32, mt, kk, lane), bfr, ah[mt]);
  }
  const float bh = bhh[n];
#pragma unroll
  for (int mt = 0; mt < 4; ++mt)
#pragma unroll
    for (int j = 0; j < 8; ++j)
      gh[(size_t)(mt * 16 + j + 8 * hh) * H3 + n] = ah[mt][j] + bh;
}

// ============================================================================
// K5: GRU gates -> h_new (f32 to output slice, swizzled bf16 to ws for K6)
// ============================================================================
__global__ void __launch_bounds__(256)
k5_gates(const float* __restrict__ gi, const float* __restrict__ gh,
         const float* __restrict__ dec_hs, unsigned short* __restrict__ hswz,
         float* __restrict__ out, size_t hnew_off) {
  const int idx = blockIdx.x * 256 + threadIdx.x;
  const int b = idx >> 10, j = idx & (Hz - 1);
  const size_t base = (size_t)b * H3 + j;
  const float gir = gi[base], ghr = gh[base];
  const float giz = gi[base + Hz], ghz = gh[base + Hz];
  const float gin = gi[base + 2 * Hz], ghn = gh[base + 2 * Hz];
  const float r = 1.0f / (1.0f + expf(-(gir + ghr)));
  const float z = 1.0f / (1.0f + expf(-(giz + ghz)));
  const float nn = tanhf(gin + r * ghn);
  const float hp = dec_hs[(size_t)b * Hz + j];
  const float hn = (1.0f - z) * nn + z * hp;
  out[hnew_off + (size_t)b * Hz + j] = hn;

  const int mt = b >> 4, m = b & 15;
  const int kk = j >> 5, ko = j & 31;
  int hh, ii;
  komap(ko, hh, ii);
  hswz[frag_off(Hz / 32, mt, kk, m + 16 * hh) + ii] = bf16_bits(hn);
}

// ============================================================================
// K6: fc_out = h_new @ fc_W^T + fc_b (64 x 32000, K=1024), memory-bound on
// fc_W (128 MB, read exactly once). One wave per N-tile (2000); each
// converted fc_W fragment feeds 4 WMMAs. A fragments pre-swizzled bf16.
// ============================================================================
__global__ void k6_fc(const unsigned short* __restrict__ hswz, const float* __restrict__ fcW,
                      const float* __restrict__ fcb, float* __restrict__ out) {
  const int lane = threadIdx.x;
  const int nt = blockIdx.x;
  const int lo = lane & 15, hh = lane >> 4;

  v8f acc[4] = {{}, {}, {}, {}};
  for (int kk = 0; kk < Hz / 32; ++kk) {
    v16bf bfr = load_frag_cvt(fcW, Hz, nt * 16, kk * 32, lane);
#pragma unroll
    for (int mt = 0; mt < 4; ++mt)
      acc[mt] = wmma_bf16(load_frag_swz(hswz, Hz / 32, mt, kk, lane), bfr, acc[mt]);
  }
  const int n = nt * 16 + lo;
  const float bias = fcb[n];
#pragma unroll
  for (int mt = 0; mt < 4; ++mt)
#pragma unroll
    for (int j = 0; j < 8; ++j)
      out[(size_t)(mt * 16 + j + 8 * hh) * Vz + n] = acc[mt][j] + bias;
}

// ============================================================================
extern "C" void kernel_launch(void* const* d_in, const int* in_sizes, int n_in,
                              void* d_out, int out_size, void* d_ws, size_t ws_size,
                              hipStream_t stream) {
  const int*   x      = (const int*)d_in[0];
  const float* dec_hs = (const float*)d_in[1];
  const float* enc    = (const float*)d_in[2];
  const float* embT   = (const float*)d_in[3];
  const float* w1W    = (const float*)d_in[4];
  const float* w1b    = (const float*)d_in[5];
  const float* w2W    = (const float*)d_in[6];
  const float* w2b    = (const float*)d_in[7];
  const float* vW     = (const float*)d_in[8];
  const float* vb     = (const float*)d_in[9];
  const float* Wih    = (const float*)d_in[10];
  const float* Whh    = (const float*)d_in[11];
  const float* bih    = (const float*)d_in[12];
  const float* bhh    = (const float*)d_in[13];
  const float* fcW    = (const float*)d_in[14];
  const float* fcb    = (const float*)d_in[15];

  unsigned short* w2s  = (unsigned short*)d_ws;            // 64*32*32*16  = 1,048,576 elems
  unsigned short* dswz = w2s + (size_t)64 * 32 * 32 * 16;  // 4*32*32*16   = 65,536
  unsigned short* uswz = dswz + (size_t)4 * 32 * 32 * 16;  // 4*40*32*16   = 81,920
  unsigned short* hswz = uswz + (size_t)4 * 40 * 32 * 16;  // 4*32*32*16   = 65,536
  float* t1_ws     = (float*)(hswz + (size_t)4 * 32 * 32 * 16);
  float* scores_ws = t1_ws + Bz * Hz;
  float* gi_ws     = scores_ws + Bz * Sz;
  float* gh_ws     = gi_ws + (size_t)Bz * H3;

  float* out = (float*)d_out;
  const size_t hnew_off = (size_t)Bz * Vz;
  const size_t attn_off = hnew_off + (size_t)Bz * Hz;

  k0_w2swz<<<(64 * 32 * 32) / 256, 256, 0, stream>>>(w2W, w2s);
  k0_decswz<<<(4 * 32 * 32) / 256, 256, 0, stream>>>(dec_hs, dswz);
  k1_t1<<<(Bz / 16) * (Hz / 16), 32, 0, stream>>>(dswz, w1W, w1b, w2b, t1_ws);
  k2_scores<<<Sz, 256, 0, stream>>>(enc, w2s, t1_ws, vW, vb, scores_ws);
  k3_softmax_ctx<<<Bz, 256, 0, stream>>>(scores_ws, enc, embT, x, uswz, out, attn_off);
  k4_gru_gemm<<<H3 / 16, 32, 0, stream>>>(uswz, dswz, Wih, Whh, bih, bhh, gi_ws, gh_ws);
  k5_gates<<<(Bz * Hz) / 256, 256, 0, stream>>>(gi_ws, gh_ws, dec_hs, hswz, out, hnew_off);
  k6_fc<<<Vz / 16, 32, 0, stream>>>(hswz, fcW, fcb, out);
}
